// LSTM_57062935495488
// MI455X (gfx1250) — matrix operand order repaired and checked by reading, not compiled, not run on hardware
//
#include <hip/hip_runtime.h>

// ---------------------------------------------------------------------------
// LSTM (T=512, B=64, D=256, H=512) for gfx1250 (MI455X).
// 512 sequential fused step-kernels (graph-capturable), each computing
// gates = [h_t, x_t] @ W^T + b with v_wmma_f32_16x16x32_bf16 (f32 accum) and
// the sigmoid/tanh cell update in-register. Weights pre-converted to bf16 in
// d_ws (3MB, L2-resident across all steps). h/C state stays f32.
// The 24-slice K loop is FULLY UNROLLED and SOFTWARE-PIPELINED: slice kk+1's
// 4 B-fragments + A-fragment are loaded into a second register set while the
// 4 WMMAs of slice kk execute, so each WMMA waits only on loads issued a full
// iteration earlier (critical since these kernels run ~1 wave/SIMD).
// ---------------------------------------------------------------------------

typedef __attribute__((ext_vector_type(16))) __bf16 v16bf;
typedef __attribute__((ext_vector_type(8)))  float  v8f;

#define T_STEPS 512
#define B_SZ    64
#define D_SZ    256
#define H_SZ    512
#define K_TOT   768          // H + D (h first, then x, matching merge order)
#define NGATES  4
#define NSLICE  (K_TOT / 32) // 24 K-slices of 32

union AVec { v16bf v; __bf16 e[16]; };
union BVec { v16bf v; uint4  q[2];  };

__device__ __forceinline__ float sig_f(float x) {
    return 1.0f / (1.0f + __expf(-x));
}
__device__ __forceinline__ float tanh_f(float x) {
    x = fminf(15.0f, fmaxf(-15.0f, x));
    float e = __expf(-2.0f * x);
    return (1.0f - e) / (1.0f + e);
}

__device__ __forceinline__ v16bf load_a_frag(const float* __restrict__ ap) {
    // 16-bit A 16x32 layout (contiguous, matches B): elems 0..15 = K(hi*16 + 0..15)
    float4 a0 = *(const float4*)(ap);
    float4 a1 = *(const float4*)(ap + 4);
    float4 a2 = *(const float4*)(ap + 8);
    float4 a3 = *(const float4*)(ap + 12);
    AVec av;
    av.e[0]  = (__bf16)a0.x; av.e[1]  = (__bf16)a0.y;
    av.e[2]  = (__bf16)a0.z; av.e[3]  = (__bf16)a0.w;
    av.e[4]  = (__bf16)a1.x; av.e[5]  = (__bf16)a1.y;
    av.e[6]  = (__bf16)a1.z; av.e[7]  = (__bf16)a1.w;
    av.e[8]  = (__bf16)a2.x; av.e[9]  = (__bf16)a2.y;
    av.e[10] = (__bf16)a2.z; av.e[11] = (__bf16)a2.w;
    av.e[12] = (__bf16)a3.x; av.e[13] = (__bf16)a3.y;
    av.e[14] = (__bf16)a3.z; av.e[15] = (__bf16)a3.w;
    return av.v;
}

__device__ __forceinline__ v16bf load_b_frag(const __bf16* __restrict__ bp) {
    // 16-bit B 32x16 layout (contiguous, matches A, not the ISA table order): lane = column N, elems 0..15 = contiguous K.
    BVec bv;
    bv.q[0] = *(const uint4*)(bp);
    bv.q[1] = *(const uint4*)(bp + 8);
    return bv.v;
}

// ---------------------------------------------------------------------------
// Kernel 1: convert W = [Wi;Wf;Wg;Wo] rows ([512,768] each) to bf16 in ws.
// ---------------------------------------------------------------------------
__global__ void __launch_bounds__(256)
LSTM_57062935495488_convw(const float* __restrict__ Wi, const float* __restrict__ Wf,
                          const float* __restrict__ Wg, const float* __restrict__ Wo,
                          __bf16* __restrict__ Wbf) {
    int idx = blockIdx.x * blockDim.x + threadIdx.x;
    const int per_gate = H_SZ * K_TOT;
    if (idx >= NGATES * per_gate) return;
    int g   = idx / per_gate;
    int rem = idx - g * per_gate;
    const float* src = (g == 0) ? Wi : (g == 1) ? Wf : (g == 2) ? Wg : Wo;
    Wbf[idx] = (__bf16)src[rem];
}

// ---------------------------------------------------------------------------
// Kernel 2: allhidden[0] = h0 ; C_tail = C0.
// ---------------------------------------------------------------------------
__global__ void __launch_bounds__(256)
LSTM_57062935495488_init(const float* __restrict__ h0, const float* __restrict__ C0,
                         float* __restrict__ allh0, float* __restrict__ Ctail) {
    int idx = blockIdx.x * blockDim.x + threadIdx.x;
    if (idx < B_SZ * H_SZ) {
        allh0[idx] = h0[idx];
        Ctail[idx] = C0[idx];
    }
}

// ---------------------------------------------------------------------------
// Kernel 3: one fused LSTM step.
//   grid = 32 blocks x 128 threads (4 wave32 per block); 128 waves total.
//   wave tile: 16 batch rows x 16 hidden cols, all 4 gates in-register.
// ---------------------------------------------------------------------------
__global__ void __launch_bounds__(128)
LSTM_57062935495488_step(const float* __restrict__ x_t,
                         const float* __restrict__ h_prev,
                         float* __restrict__ h_out,
                         float* __restrict__ C,
                         const __bf16* __restrict__ Wbf,
                         const float* __restrict__ bi, const float* __restrict__ bff,
                         const float* __restrict__ bg, const float* __restrict__ bo) {
    const int lane = threadIdx.x & 31;
    const int wave = threadIdx.x >> 5;
    const int tile = blockIdx.x * 4 + wave;      // 0..127
    const int brow = tile & 3;                   // B/16 = 4 tiles
    const int hcol = tile >> 2;                  // H/16 = 32 tiles

    const int l16    = lane & 15;
    const int hi     = lane >> 4;                // half-wave select
    const int bglobA = brow * 16 + l16;          // A row (batch) for this lane
    const int nglob  = hcol * 16 + l16;          // C/D column (hidden unit)
    const int mbase  = hi * 8;                   // C/D row base for this lane

    // Per-gate weight row base, pre-offset by the lane's K phase (hi*16).
    const __bf16* w0 = Wbf + (size_t)(0 * H_SZ + nglob) * K_TOT + hi * 16;
    const __bf16* w1 = Wbf + (size_t)(1 * H_SZ + nglob) * K_TOT + hi * 16;
    const __bf16* w2 = Wbf + (size_t)(2 * H_SZ + nglob) * K_TOT + hi * 16;
    const __bf16* w3 = Wbf + (size_t)(3 * H_SZ + nglob) * K_TOT + hi * 16;

    const float* hrow = h_prev + (size_t)bglobA * H_SZ + hi * 16;
    const float* xrow = x_t    + (size_t)bglobA * D_SZ + hi * 16;

    // Hoisted epilogue operands: their latency hides under the whole GEMM.
    const float bi_n = bi[nglob];
    const float bf_n = bff[nglob];
    const float bg_n = bg[nglob];
    const float bo_n = bo[nglob];
    float c_old[8];
    #pragma unroll
    for (int r = 0; r < 8; ++r)
        c_old[r] = C[(size_t)(brow * 16 + mbase + r) * H_SZ + nglob];

    v8f acc0 = {}, acc1 = {}, acc2 = {}, acc3 = {};

    // ---- Software-pipelined, fully-unrolled K loop (24 slices of 32) ----
    // Prologue: load slice 0.
    v16bf cb0 = load_b_frag(w0);
    v16bf cb1 = load_b_frag(w1);
    v16bf cb2 = load_b_frag(w2);
    v16bf cb3 = load_b_frag(w3);
    v16bf ca  = load_a_frag(hrow);

    #pragma unroll
    for (int kk = 0; kk < NSLICE - 1; ++kk) {
        const int Kn = (kk + 1) * 32;
        // Issue slice kk+1 loads into the second register set (the compile-
        // time h/x select vanishes under full unroll).
        v16bf nb0 = load_b_frag(w0 + Kn);
        v16bf nb1 = load_b_frag(w1 + Kn);
        v16bf nb2 = load_b_frag(w2 + Kn);
        v16bf nb3 = load_b_frag(w3 + Kn);
        const float* an = (Kn < H_SZ) ? (hrow + Kn) : (xrow + (Kn - H_SZ));
        v16bf na = load_a_frag(an);
        // Consume slice kk while slice kk+1 is in flight.
        acc0 = __builtin_amdgcn_wmma_f32_16x16x32_bf16(false, ca, false, cb0, (short)0, acc0, false, false);
        acc1 = __builtin_amdgcn_wmma_f32_16x16x32_bf16(false, ca, false, cb1, (short)0, acc1, false, false);
        acc2 = __builtin_amdgcn_wmma_f32_16x16x32_bf16(false, ca, false, cb2, (short)0, acc2, false, false);
        acc3 = __builtin_amdgcn_wmma_f32_16x16x32_bf16(false, ca, false, cb3, (short)0, acc3, false, false);
        cb0 = nb0; cb1 = nb1; cb2 = nb2; cb3 = nb3; ca = na;
    }
    // Epilogue slice.
    acc0 = __builtin_amdgcn_wmma_f32_16x16x32_bf16(false, ca, false, cb0, (short)0, acc0, false, false);
    acc1 = __builtin_amdgcn_wmma_f32_16x16x32_bf16(false, ca, false, cb1, (short)0, acc1, false, false);
    acc2 = __builtin_amdgcn_wmma_f32_16x16x32_bf16(false, ca, false, cb2, (short)0, acc2, false, false);
    acc3 = __builtin_amdgcn_wmma_f32_16x16x32_bf16(false, ca, false, cb3, (short)0, acc3, false, false);

    // ---- Fused LSTM cell update (C/D layout: lane -> N, VGPR r -> M) ----
    #pragma unroll
    for (int r = 0; r < 8; ++r) {
        const int bglob = brow * 16 + mbase + r;
        const size_t idx = (size_t)bglob * H_SZ + nglob;
        float iv = sig_f(acc0[r] + bi_n);
        float fv = sig_f(acc1[r] + bf_n);
        float gv = tanh_f(acc2[r] + bg_n);
        float ov = sig_f(acc3[r] + bo_n);
        float cn = fmaf(fv, c_old[r], iv * gv);
        C[idx]     = cn;
        h_out[idx] = ov * tanh_f(cn);
    }
}

// ---------------------------------------------------------------------------
// Host launcher
// ---------------------------------------------------------------------------
extern "C" void kernel_launch(void* const* d_in, const int* in_sizes, int n_in,
                              void* d_out, int out_size, void* d_ws, size_t ws_size,
                              hipStream_t stream) {
    const float* seqs = (const float*)d_in[0];   // [T, B, D]
    const float* h0   = (const float*)d_in[1];   // [B, H]
    const float* C0   = (const float*)d_in[2];   // [B, H]
    const float* Wi   = (const float*)d_in[3];   // [H, D+H]
    const float* bi   = (const float*)d_in[4];
    const float* Wf   = (const float*)d_in[5];
    const float* bf_  = (const float*)d_in[6];
    const float* Wg   = (const float*)d_in[7];
    const float* bg   = (const float*)d_in[8];
    const float* Wo   = (const float*)d_in[9];
    const float* bo   = (const float*)d_in[10];

    float* allh  = (float*)d_out;                              // [T+1, B, H]
    float* Ctail = allh + (size_t)(T_STEPS + 1) * B_SZ * H_SZ; // [B, H] (cT)

    __bf16* Wbf = (__bf16*)d_ws;   // [4, H, K_TOT] bf16 = 3 MB

    // 1) Convert weights to bf16 (L2-resident for all 512 steps).
    {
        int n = NGATES * H_SZ * K_TOT;
        LSTM_57062935495488_convw<<<(n + 255) / 256, 256, 0, stream>>>(Wi, Wf, Wg, Wo, Wbf);
    }
    // 2) Initialize allhidden[0] and the cell-state tail of d_out.
    {
        int n = B_SZ * H_SZ;
        LSTM_57062935495488_init<<<(n + 255) / 256, 256, 0, stream>>>(h0, C0, allh, Ctail);
    }
    // 3) 512 sequential fused step kernels (stream order = recurrence order).
    for (int t = 0; t < T_STEPS; ++t) {
        LSTM_57062935495488_step<<<32, 128, 0, stream>>>(
            seqs + (size_t)t * B_SZ * D_SZ,
            allh + (size_t)t * B_SZ * H_SZ,
            allh + (size_t)(t + 1) * B_SZ * H_SZ,
            Ctail, Wbf, bi, bf_, bg, bo);
    }
}
